// QuantumAttention_65481071407673
// MI455X (gfx1250) — compile-verified, matrix operand order checked by reading, not run on hardware
//
#include <hip/hip_runtime.h>
#include <hip/hip_bf16.h>

#define BATCH 2
#define SEQ   1024
#define EMB   256
#define HEADS 64
#define DKK   4

typedef __attribute__((ext_vector_type(16))) _Float16 v16h;
typedef __attribute__((ext_vector_type(8)))  float    v8f;
typedef __attribute__((ext_vector_type(2)))  float    v2f;

#if defined(__has_builtin)
#if __has_builtin(__builtin_amdgcn_wmma_f32_16x16x4_f32)
#define HAVE_WMMA_F32X4 1
#endif
#endif

// ---------------------------------------------------------------------------
// Kernel A: fused QKV projection + quantum circuit epilogue.
// Computes D = W_tile(A, 16 embed rows x 32 k) x x^T_tile(B, 32 k x 16 seq)
// so each lane's 8 f32 accumulators hold 8 consecutive EMBED dims of one
// sequence position = exactly 2 complete 4-wire head groups -> the quantum
// transform (z_w = cos(p_w)*cos(ang_w + p_w); outputs = XOR-set products of z)
// is pure per-lane math. Output layout (B,H,S,4) f32.
// ---------------------------------------------------------------------------
__global__ __launch_bounds__(256) void qa_qkv_quantum(
    const float* __restrict__ x,
    const float* __restrict__ Wq, const float* __restrict__ Wk,
    const float* __restrict__ Wv, const float* __restrict__ params,
    float* __restrict__ qp, float* __restrict__ kp, float* __restrict__ vp)
{
  const int lane = threadIdx.x & 31;
  const int wave = threadIdx.x >> 5;
  const int lo = lane & 15, hi = lane >> 4;

  const int tile = blockIdx.x * 8 + wave;   // 3 mats * 16 n-tiles * 128 m-tiles = 6144
  const int mat  = tile >> 11;              // / 2048
  const int rem  = tile & 2047;
  const int nt   = rem >> 7;                // embed tile (16)
  const int mt   = rem & 127;               // seq tile (128)

  const float* W    = (mat == 0) ? Wq : (mat == 1) ? Wk : Wv;
  float*       outp = (mat == 0) ? qp : (mat == 1) ? kp : vp;

  const float* wrow = W + (nt * 16 + lo) * EMB;  // A operand: row = embed dim
  const float* xrow = x + (mt * 16 + lo) * EMB;  // B operand: column = seq pos

  v8f c = {};
  #pragma unroll
  for (int ks = 0; ks < EMB; ks += 32) {
    // 16-bit A layout: lane holds K = {hi*8..hi*8+7} and {16+hi*8..16+hi*8+7};
    // B mirrored (lane = column n, same K chunking). Both are K-contiguous in
    // memory (x @ W.T is a TN GEMM), so plain b128 loads + f32->f16 converts.
    const float* wa = wrow + ks + hi * 8;
    const float* xa = xrow + ks + hi * 8;
    const float4 w0 = *(const float4*)(wa);
    const float4 w1 = *(const float4*)(wa + 4);
    const float4 w2 = *(const float4*)(wa + 16);
    const float4 w3 = *(const float4*)(wa + 20);
    const float4 x0 = *(const float4*)(xa);
    const float4 x1 = *(const float4*)(xa + 4);
    const float4 x2 = *(const float4*)(xa + 16);
    const float4 x3 = *(const float4*)(xa + 20);
    v16h a, b;
    a[0]=(_Float16)w0.x; a[1]=(_Float16)w0.y; a[2]=(_Float16)w0.z; a[3]=(_Float16)w0.w;
    a[4]=(_Float16)w1.x; a[5]=(_Float16)w1.y; a[6]=(_Float16)w1.z; a[7]=(_Float16)w1.w;
    a[8]=(_Float16)w2.x; a[9]=(_Float16)w2.y; a[10]=(_Float16)w2.z; a[11]=(_Float16)w2.w;
    a[12]=(_Float16)w3.x; a[13]=(_Float16)w3.y; a[14]=(_Float16)w3.z; a[15]=(_Float16)w3.w;
    b[0]=(_Float16)x0.x; b[1]=(_Float16)x0.y; b[2]=(_Float16)x0.z; b[3]=(_Float16)x0.w;
    b[4]=(_Float16)x1.x; b[5]=(_Float16)x1.y; b[6]=(_Float16)x1.z; b[7]=(_Float16)x1.w;
    b[8]=(_Float16)x2.x; b[9]=(_Float16)x2.y; b[10]=(_Float16)x2.z; b[11]=(_Float16)x2.w;
    b[12]=(_Float16)x3.x; b[13]=(_Float16)x3.y; b[14]=(_Float16)x3.z; b[15]=(_Float16)x3.w;
    c = __builtin_amdgcn_wmma_f32_16x16x32_f16(false, a, false, b, (short)0, c,
                                               false, false);
  }

  // Quantum epilogue. c[j] = proj[seq = mt*16+lo, embed = nt*16 + 8*hi + j].
  const float p0 = params[0], p1 = params[1], p2 = params[2], p3 = params[3];
  const float cp0 = __cosf(p0), cp1 = __cosf(p1), cp2 = __cosf(p2), cp3 = __cosf(p3);
  const int sg = mt * 16 + lo;        // global row in (B*S)
  const int bb = sg >> 10;            // batch
  const int ss = sg & 1023;           // seq pos
  #pragma unroll
  for (int g = 0; g < 2; ++g) {
    const float z0 = cp0 * __cosf(c[4 * g + 0] + p0);
    const float z1 = cp1 * __cosf(c[4 * g + 1] + p1);
    const float z2 = cp2 * __cosf(c[4 * g + 2] + p2);
    const float z3 = cp3 * __cosf(c[4 * g + 3] + p3);
    const float o1 = z0 * z1;
    const float o2 = o1 * z2;
    const float o3 = o2 * z3;
    const float o0 = z1 * z2 * z3;
    const int hcol = nt * 16 + hi * 8 + g * 4;   // embed base of this head group
    const int head = hcol >> 2;
    *(float4*)(outp + (((bb * HEADS + head) * SEQ + ss) << 2)) =
        make_float4(o0, o1, o2, o3);
  }
}

// ---------------------------------------------------------------------------
// Kernel B: flash attention, one workgroup per (b,h). dk=4 == WMMA K for
// V_WMMA_F32_16X16X4_F32: one WMMA per 16x16 score tile, exact f32.
// K/V head staged in LDS (32 KB of the 320 KB WGP pool). Row softmax stats
// reduced across 16-lane halves with shfl_xor (halves own disjoint rows).
// ---------------------------------------------------------------------------
__global__ __launch_bounds__(256) void qa_attention(
    const float* __restrict__ qp, const float* __restrict__ kp,
    const float* __restrict__ vp, float* __restrict__ out)
{
  __shared__ float ksm[SEQ * 4];
  __shared__ float vsm[SEQ * 4];
  __shared__ float red[8][16][16][4];   // per-wave ctx-partial reduction

  const int bh = blockIdx.x;            // 0..127
  const float* qh = qp + bh * SEQ * 4;
  const float* kh = kp + bh * SEQ * 4;
  const float* vh = vp + bh * SEQ * 4;

  for (int i = threadIdx.x; i < SEQ; i += 256) {
    *(float4*)&ksm[i * 4] = *(const float4*)(kh + i * 4);
    *(float4*)&vsm[i * 4] = *(const float4*)(vh + i * 4);
  }
  __syncthreads();

  const int lane = threadIdx.x & 31, wave = threadIdx.x >> 5;
  const int lo = lane & 15, hi = lane >> 4;
  const int bb = bh >> 6, head = bh & 63;

  for (int qt = wave; qt < 64; qt += 8) {   // 8 q-tiles per wave, uniform
#if HAVE_WMMA_F32X4
    // A operand (16x4 f32): lane holds row M=lo, K = {2*hi, 2*hi+1}.
    v2f aq;
    aq[0] = 0.5f * qh[(qt * 16 + lo) * 4 + hi * 2 + 0];  // fold 1/sqrt(dk)
    aq[1] = 0.5f * qh[(qt * 16 + lo) * 4 + hi * 2 + 1];
#else
    float4 qr[8];
    #pragma unroll
    for (int j = 0; j < 8; ++j) {
      float4 q4 = *(const float4*)(qh + (qt * 16 + j + 8 * hi) * 4);
      qr[j] = make_float4(0.5f * q4.x, 0.5f * q4.y, 0.5f * q4.z, 0.5f * q4.w);
    }
#endif
    float run_max[8], run_sum[8];
    float4 ctxp[8];
    #pragma unroll
    for (int j = 0; j < 8; ++j) {
      run_max[j] = -1e30f; run_sum[j] = 0.0f;
      ctxp[j] = make_float4(0.f, 0.f, 0.f, 0.f);
    }

    for (int ct = 0; ct < 64; ++ct) {
      const int ncol = ct * 16 + lo;
      const float4 vcol = *(const float4*)&vsm[ncol * 4];
#if HAVE_WMMA_F32X4
      v2f bk;
      bk[0] = ksm[ncol * 4 + hi * 2 + 0];
      bk[1] = ksm[ncol * 4 + hi * 2 + 1];
      v8f sc = {};
      sc = __builtin_amdgcn_wmma_f32_16x16x4_f32(false, aq, false, bk,
                                                 (short)0, sc, false, false);
#else
      const float4 kc = *(const float4*)&ksm[ncol * 4];
      float sc[8];
      #pragma unroll
      for (int j = 0; j < 8; ++j)
        sc[j] = qr[j].x * kc.x + qr[j].y * kc.y + qr[j].z * kc.z + qr[j].w * kc.w;
#endif
      // Online softmax: rows j+8*hi live in VGPR j, columns across the 16
      // lanes of this half -> xor masks 1,2,4,8 reduce each half separately.
      #pragma unroll
      for (int j = 0; j < 8; ++j) {
        const float s = sc[j];
        float mx = s;
        mx = fmaxf(mx, __shfl_xor(mx, 1));
        mx = fmaxf(mx, __shfl_xor(mx, 2));
        mx = fmaxf(mx, __shfl_xor(mx, 4));
        mx = fmaxf(mx, __shfl_xor(mx, 8));
        const float nm    = fmaxf(run_max[j], mx);
        const float scale = __expf(run_max[j] - nm);
        const float pr    = __expf(s - nm);
        float sum = pr;
        sum += __shfl_xor(sum, 1);
        sum += __shfl_xor(sum, 2);
        sum += __shfl_xor(sum, 4);
        sum += __shfl_xor(sum, 8);
        run_sum[j] = run_sum[j] * scale + sum;
        run_max[j] = nm;
        ctxp[j].x = ctxp[j].x * scale + pr * vcol.x;
        ctxp[j].y = ctxp[j].y * scale + pr * vcol.y;
        ctxp[j].z = ctxp[j].z * scale + pr * vcol.z;
        ctxp[j].w = ctxp[j].w * scale + pr * vcol.w;
      }
    }

    // Normalize, dump per-lane partials, reduce across the 16 columns in LDS.
    #pragma unroll
    for (int j = 0; j < 8; ++j) {
      const float inv = 1.0f / run_sum[j];
      const int r = j + 8 * hi;
      *(float4*)&red[wave][r][lo][0] =
          make_float4(ctxp[j].x * inv, ctxp[j].y * inv,
                      ctxp[j].z * inv, ctxp[j].w * inv);
    }
    __syncthreads();
    #pragma unroll
    for (int t = 0; t < 2; ++t) {
      const int o = lane * 2 + t;     // 64 outputs: 16 rows x 4 dims
      const int r = o >> 2, d = o & 3;
      float acc = 0.0f;
      #pragma unroll
      for (int cc = 0; cc < 16; ++cc) acc += red[wave][r][cc][d];
      const int srow = qt * 16 + r;
      out[(bb * SEQ + srow) * EMB + head * 4 + d] = acc;   // (B,S,E) directly
    }
    __syncthreads();
  }
}

extern "C" void kernel_launch(void* const* d_in, const int* in_sizes, int n_in,
                              void* d_out, int out_size, void* d_ws, size_t ws_size,
                              hipStream_t stream) {
  const float* x      = (const float*)d_in[0];
  const float* Wq     = (const float*)d_in[1];
  const float* Wk     = (const float*)d_in[2];
  const float* Wv     = (const float*)d_in[3];
  const float* params = (const float*)d_in[4];
  float* out = (float*)d_out;

  const size_t headElems = (size_t)BATCH * HEADS * SEQ * DKK;  // 524288 floats
  float* qp = (float*)d_ws;
  float* kp = qp + headElems;
  float* vp = kp + headElems;

  // 6144 wave-tiles / 8 waves per block = 768 blocks
  qa_qkv_quantum<<<768, 256, 0, stream>>>(x, Wq, Wk, Wv, params, qp, kp, vp);
  // one workgroup per (b, h)
  qa_attention<<<BATCH * HEADS, 256, 0, stream>>>(qp, kp, vp, out);
}
